// NeuralODE_29300266893864
// MI455X (gfx1250) — matrix operand order repaired and checked by reading, not compiled, not run on hardware
//
#include <hip/hip_runtime.h>

typedef _Float16 half_t;
typedef __attribute__((ext_vector_type(16))) _Float16 v16h;
typedef __attribute__((ext_vector_type(8)))  _Float16 v8h;
typedef __attribute__((ext_vector_type(8)))  float    v8f;

#define LDW   264      // padded LDS row stride in halves (16B aligned, bank shift 4)
#define HID   256
#define ODIM  64
#define TSTEPS 100
#define BATCHN 4096
#define ROWS_PER_BLOCK 32

// A fragment: 16x32 f16 tile, row-major in LDS with stride LDW.
// lanes 0-15: M=lane, K pairs 0..7 and 16..23 ; lanes 16-31: M=lane-16, K 8..15 and 24..31
__device__ __forceinline__ v16h ld_fragA(const half_t* p, int mlane, int khalf) {
  const half_t* rp = p + mlane * LDW + khalf;          // khalf = (lane>>4)*8
  v8h a = *(const v8h*)(rp);
  v8h b = *(const v8h*)(rp + 16);
  v16h r;
#pragma unroll
  for (int i = 0; i < 8; ++i) { r[i] = a[i]; r[8 + i] = b[i]; }
  return r;
}

// B fragment: 32x16 (KxN) where B[k][n] = W[n][k], W row-major [n][k] stride LDW.
// lanes 0-15: N=lane, K 0..7 and 16..23 ; lanes 16-31: N=lane-16, K 8..15 and 24..31
__device__ __forceinline__ v16h ld_fragB(const half_t* p, int nlane, int k16) {
  const half_t* rp = p + nlane * LDW + (k16 >> 1);     // k16 = (lane>>4)*16
  v8h a = *(const v8h*)(rp);
  v8h b = *(const v8h*)(rp + 16);
  v16h r;
#pragma unroll
  for (int i = 0; i < 8; ++i) { r[i] = a[i]; r[8 + i] = b[i]; }
  return r;
}

// acc[4] tiles: out[m_off..+16, n_base + ns*16 ..] += A(16xK) x W^T ; K = KS*32
template <int KS>
__device__ __forceinline__ void gemm_tile(const half_t* __restrict__ sA,
                                          const half_t* __restrict__ wB,
                                          int n_base, int mlane, int khalf, int k16,
                                          v8f acc[4]) {
#pragma unroll
  for (int ks = 0; ks < KS; ++ks) {
    v16h a = ld_fragA(sA + ks * 32, mlane, khalf);
#pragma unroll
    for (int ns = 0; ns < 4; ++ns) {
      v16h b = ld_fragB(wB + (n_base + ns * 16) * LDW + ks * 32, mlane, k16);
      acc[ns] = __builtin_amdgcn_wmma_f32_16x16x32_f16(
          false, a, false, b, (short)0, acc[ns], false, false);
    }
  }
}

__global__ __launch_bounds__(256) void node_rk4_kernel(
    const float* __restrict__ x,    const float* __restrict__ tsp,
    const float* __restrict__ Win,  const float* __restrict__ bin,
    const float* __restrict__ W1,   const float* __restrict__ b1,
    const float* __restrict__ W2,   const float* __restrict__ b2,
    const float* __restrict__ Wout, const float* __restrict__ bout,
    float* __restrict__ out) {
  extern __shared__ __align__(16) half_t smem[];
  half_t* w1s = smem;                       // 256 x LDW
  half_t* w2s = w1s + HID * LDW;            // 256 x LDW
  half_t* wos = w2s + HID * LDW;            // 64  x LDW
  half_t* sb  = wos + ODIM * LDW;           // 32  x LDW (state / intermediate)

  const int tid   = threadIdx.x;
  const int lane  = tid & 31;
  const int wv    = tid >> 5;
  const int m_sub = wv & 1;                 // 16-row block
  const int n_q   = wv >> 1;                // 64-col block
  const int n_base = n_q * 64;
  const int mlane = lane & 15;
  const int hi    = lane >> 4;
  const int khalf = hi * 8;
  const int k16   = hi * 16;
  const int ncol  = mlane;
  const int m_off = m_sub * 16;
  const int rbase = blockIdx.x * ROWS_PER_BLOCK;

  // ---- stage x tile (32x64) and Win (256x64, temporarily in w1s) as f16 ----
  for (int i = tid; i < ROWS_PER_BLOCK * ODIM; i += 256) {
    int r = i >> 6, c = i & 63;
    sb[r * LDW + c] = (half_t)x[(size_t)(rbase + r) * ODIM + c];
  }
  for (int i = tid; i < HID * ODIM; i += 256) {
    int r = i >> 6, c = i & 63;
    w1s[r * LDW + c] = (half_t)Win[i];
  }
  __syncthreads();

  // per-lane bias values (fixed column per lane for the whole kernel)
  float b1v[4], b2v[4];
#pragma unroll
  for (int ns = 0; ns < 4; ++ns) {
    int n = n_base + ns * 16 + ncol;
    b1v[ns] = b1[n];
    b2v[ns] = b2[n];
  }
  const int nout = n_q * 16 + ncol;
  const float boutv = bout[nout];

  // ---- h0 = x @ Win^T + bin ----
  v8f h[4], hs[4];
  {
    v8f acc[4] = {};
    gemm_tile<2>(sb + m_off * LDW, w1s, n_base, mlane, khalf, k16, acc);
#pragma unroll
    for (int ns = 0; ns < 4; ++ns) {
      float bv = bin[n_base + ns * 16 + ncol];
#pragma unroll
      for (int r = 0; r < 8; ++r) h[ns][r] = acc[ns][r] + bv;
    }
  }
  __syncthreads();

  // ---- stage resident weights W1, W2, Wout as f16 ----
  for (int i = tid; i < HID * HID; i += 256) {
    int r = i >> 8, c = i & 255;
    w1s[r * LDW + c] = (half_t)W1[i];
    w2s[r * LDW + c] = (half_t)W2[i];
  }
  for (int i = tid; i < ODIM * HID; i += 256) {
    int r = i >> 8, c = i & 255;
    wos[r * LDW + c] = (half_t)Wout[i];
  }
  // write h0 (f16) into state buffer
#pragma unroll
  for (int ns = 0; ns < 4; ++ns)
#pragma unroll
    for (int r = 0; r < 8; ++r) {
      int m = m_off + r + hi * 8;
      sb[m * LDW + n_base + ns * 16 + ncol] = (half_t)h[ns][r];
    }
  __syncthreads();

  const float c_in[3] = {0.5f, 0.5f, 1.0f};
  const float c_w[4]  = {1.f / 6.f, 1.f / 3.f, 1.f / 3.f, 1.f / 6.f};

#pragma unroll 1
  for (int t = 0; t < TSTEPS; ++t) {
    // ---- out[t] = h @ Wout^T + bout   (one 16x16 tile per wave) ----
    {
      v8f po = {};
#pragma unroll
      for (int ks = 0; ks < 8; ++ks) {
        v16h a = ld_fragA(sb + m_off * LDW + ks * 32, mlane, khalf);
        v16h b = ld_fragB(wos + (n_q * 16) * LDW + ks * 32, mlane, k16);
        po = __builtin_amdgcn_wmma_f32_16x16x32_f16(
            false, a, false, b, (short)0, po, false, false);
      }
#pragma unroll
      for (int r = 0; r < 8; ++r) {
        int m = m_off + r + hi * 8;
        out[((size_t)t * BATCHN + rbase + m) * ODIM + nout] = po[r] + boutv;
      }
    }
    if (t == TSTEPS - 1) break;

    const float dt = tsp[t + 1] - tsp[t];
#pragma unroll
    for (int ns = 0; ns < 4; ++ns) hs[ns] = h[ns];

#pragma unroll 1
    for (int st = 0; st < 4; ++st) {
      // layer 1: a1 = s @ W1^T
      v8f a1[4] = {};
      gemm_tile<8>(sb + m_off * LDW, w1s, n_base, mlane, khalf, k16, a1);
      __syncthreads();
      // u = relu(a1 + b1) -> sb (each element owned by exactly one wave)
#pragma unroll
      for (int ns = 0; ns < 4; ++ns)
#pragma unroll
        for (int r = 0; r < 8; ++r) {
          float v = a1[ns][r] + b1v[ns];
          v = v > 0.f ? v : 0.f;
          int m = m_off + r + hi * 8;
          sb[m * LDW + n_base + ns * 16 + ncol] = (half_t)v;
        }
      __syncthreads();
      // layer 2: k = u @ W2^T
      v8f kk[4] = {};
      gemm_tile<8>(sb + m_off * LDW, w2s, n_base, mlane, khalf, k16, kk);
      __syncthreads();   // everyone done reading u before overwriting sb
      // RK4 combine + next stage input
      const float wgt = dt * c_w[st];
#pragma unroll
      for (int ns = 0; ns < 4; ++ns) {
#pragma unroll
        for (int r = 0; r < 8; ++r) {
          float kv = kk[ns][r] + b2v[ns];
          hs[ns][r] += wgt * kv;
          float snew;
          if (st < 3) {
            snew = h[ns][r] + (dt * c_in[st]) * kv;
          } else {
            h[ns][r] = hs[ns][r];
            snew = hs[ns][r];
          }
          int m = m_off + r + hi * 8;
          sb[m * LDW + n_base + ns * 16 + ncol] = (half_t)snew;
        }
      }
      __syncthreads();
    }
  }
}

extern "C" void kernel_launch(void* const* d_in, const int* in_sizes, int n_in,
                              void* d_out, int out_size, void* d_ws, size_t ws_size,
                              hipStream_t stream) {
  (void)in_sizes; (void)n_in; (void)out_size; (void)d_ws; (void)ws_size;
  const float* x    = (const float*)d_in[0];
  const float* tsp  = (const float*)d_in[1];
  const float* Win  = (const float*)d_in[2];
  const float* bin  = (const float*)d_in[3];
  const float* W1   = (const float*)d_in[4];
  const float* b1   = (const float*)d_in[5];
  const float* W2   = (const float*)d_in[6];
  const float* b2   = (const float*)d_in[7];
  const float* Wout = (const float*)d_in[8];
  const float* bout = (const float*)d_in[9];

  const size_t shmem =
      (size_t)(HID * LDW + HID * LDW + ODIM * LDW + ROWS_PER_BLOCK * LDW) * sizeof(half_t);

  node_rk4_kernel<<<BATCHN / ROWS_PER_BLOCK, 256, shmem, stream>>>(
      x, tsp, Win, bin, W1, b1, W2, b2, Wout, bout, (float*)d_out);
}
